// MultiheadAttention_52037823758844
// MI455X (gfx1250) — matrix operand order repaired and checked by reading, not compiled, not run on hardware
//
#include <hip/hip_runtime.h>
#include <hip/hip_bf16.h>
#include <stdint.h>

// ---------------------------------------------------------------------------
// MI455X (gfx1250) multihead attention, bf16 WMMA pipeline.
//   S=2048, B=4, E=1024, H=16, D=64.  ~137 GFLOP -> compute bound, so all
//   matmuls use v_wmma_f32_16x16x32_bf16 (f32 accumulate).  Workspace (d_ws)
//   holds bf16 Q/K/V in head layout + bf16 attention output: 64 MB total,
//   which fits entirely in the 192 MB L2.
// ---------------------------------------------------------------------------

typedef __attribute__((ext_vector_type(16))) __bf16 v16bf;
typedef __attribute__((ext_vector_type(8)))  __bf16 v8bf;
typedef __attribute__((ext_vector_type(8)))  float  v8f;

static constexpr int S_ = 2048, B_ = 4, E_ = 1024, H_ = 16, D_ = 64;
static constexpr int N_QKV = 3 * E_;                 // 3072
static constexpr size_t HEAD_ELEMS = (size_t)B_ * H_ * S_ * D_;   // 8,388,608

#define WMMA_BF16(a, b, c) \
  __builtin_amdgcn_wmma_f32_16x16x32_bf16(false, (a), false, (b), (short)0, (c), false, false)

// Build a 16-wide bf16 fragment from two contiguous 8-element (16B) chunks.
// A-fragment (16x32, lane: row=lane&15, kh=lane>>4):
//   elem i<8  -> A[row][kh*16 + i]       (chunk p0)
//   elem i>=8 -> A[row][kh*16 + i]       (chunk p1 = p0 + 8)
// B-fragment (32x16, lane: col=lane&15, kh=lane>>4):
//   elem i -> B[kh*16 + i][col]  == rowmajor-in-k slab, chunks p0, p0+8
__device__ __forceinline__ v16bf make_frag(const __bf16* p0, const __bf16* p1) {
  v8bf lo = *(const v8bf*)p0;
  v8bf hi = *(const v8bf*)p1;
  v16bf r;
#pragma unroll
  for (int i = 0; i < 8; ++i) { r[i] = lo[i]; r[i + 8] = hi[i]; }
  return r;
}

__device__ __forceinline__ void cvt16_f32_to_bf16(__bf16* dst, const float4* s) {
  float4 f0 = s[0], f1 = s[1], f2 = s[2], f3 = s[3];
  v8bf lo, hi;
  lo[0] = (__bf16)f0.x; lo[1] = (__bf16)f0.y; lo[2] = (__bf16)f0.z; lo[3] = (__bf16)f0.w;
  lo[4] = (__bf16)f1.x; lo[5] = (__bf16)f1.y; lo[6] = (__bf16)f1.z; lo[7] = (__bf16)f1.w;
  hi[0] = (__bf16)f2.x; hi[1] = (__bf16)f2.y; hi[2] = (__bf16)f2.z; hi[3] = (__bf16)f2.w;
  hi[4] = (__bf16)f3.x; hi[5] = (__bf16)f3.y; hi[6] = (__bf16)f3.z; hi[7] = (__bf16)f3.w;
  *(v8bf*)dst = lo;
  *(v8bf*)(dst + 8) = hi;
}

// ---- async global->LDS path (gfx1250), guarded so missing builtins fall back
#if defined(__HIP_DEVICE_COMPILE__) && \
    __has_builtin(__builtin_amdgcn_global_load_async_to_lds_b128) && \
    __has_builtin(__builtin_amdgcn_s_wait_asynccnt)
#define USE_ASYNC_LDS 1
// Parameter types per hipcc diagnostic: pointee is int __vector_size(16),
// AS1 (printed "__device__") for the global source, AS3 for the LDS side.
typedef int v4i_vs __attribute__((vector_size(16)));
typedef __attribute__((address_space(1))) v4i_vs as1_v4i;
typedef __attribute__((address_space(3))) v4i_vs as3_v4i;
#endif

__device__ __forceinline__ void cp16_g2l(void* lds_dst, const void* gsrc) {
#if defined(USE_ASYNC_LDS)
  __builtin_amdgcn_global_load_async_to_lds_b128(
      (as1_v4i*)(uintptr_t)gsrc, (as3_v4i*)(uintptr_t)lds_dst, 0, 0);
#else
  *(uint4*)lds_dst = *(const uint4*)gsrc;
#endif
}
__device__ __forceinline__ void cp_wait_all() {
#if defined(USE_ASYNC_LDS)
  __builtin_amdgcn_s_wait_asynccnt(0);
#endif
}

// ---------------------------------------------------------------------------
// Kernel 1: qkv = x @ Wqkv^T + bias, cast to bf16, scatter into head layout.
//   x: [8192][1024] f32 (row m = s*B + b), W: [3072][1024] f32.
//   Output n -> which(q/k/v), head h=(n%1024)/64, d=n%64; q scaled by
//   0.125 * log2(e) so attention softmax runs in exp2 domain.
//   WG = 256 thr (8 waves), tile 128x128, K step 32.
// ---------------------------------------------------------------------------
__global__ __launch_bounds__(256)
void mha_qkv_proj_kernel(const float* __restrict__ X, const float* __restrict__ W,
                         const float* __restrict__ bias,
                         __bf16* __restrict__ qh, __bf16* __restrict__ kh,
                         __bf16* __restrict__ vh) {
  __shared__ __align__(16) __bf16 As[128][40];   // [row][k], 80B rows
  __shared__ __align__(16) __bf16 Bs[128][40];   // [col n][k] == W rows

  const int tid = threadIdx.x;
  const int lane = tid & 31, wv = tid >> 5;
  const int l15 = lane & 15, lh = lane >> 4;
  const int m0 = blockIdx.y * 128;
  const int n0 = blockIdx.x * 128;

  const int ldrow = tid >> 1;    // 0..127, one LDS row per 2 threads
  const int ldh   = tid & 1;     // 16-element half of the 32-wide k slab
  const float* xsrc = X + (size_t)(m0 + ldrow) * E_ + ldh * 16;
  const float* wsrc = W + (size_t)(n0 + ldrow) * E_ + ldh * 16;

  v8f acc[8] = {};

  for (int kb = 0; kb < E_; kb += 32) {
    cvt16_f32_to_bf16(&As[ldrow][ldh * 16], (const float4*)xsrc);
    cvt16_f32_to_bf16(&Bs[ldrow][ldh * 16], (const float4*)wsrc);
    xsrc += 32; wsrc += 32;
    if (kb + 32 < E_) {            // lowers to global_prefetch_b8
      __builtin_prefetch(xsrc, 0, 1);
      __builtin_prefetch(wsrc, 0, 1);
    }
    __syncthreads();

    v16bf afrag = make_frag(&As[wv * 16 + l15][lh * 16],
                            &As[wv * 16 + l15][lh * 16 + 8]);
#pragma unroll
    for (int c = 0; c < 8; ++c) {
      v16bf bfrag = make_frag(&Bs[c * 16 + l15][lh * 16],
                              &Bs[c * 16 + l15][lh * 16 + 8]);
      acc[c] = WMMA_BF16(afrag, bfrag, acc[c]);
    }
    __syncthreads();
  }

  const float qscale = 0.125f * 1.44269504088896340736f;  // 1/sqrt(64)*log2(e)
#pragma unroll
  for (int c = 0; c < 8; ++c) {
    const int n = n0 + c * 16 + l15;
    const int which = n >> 10;
    const int e = n & 1023;
    const int h = e >> 6, d = e & 63;
    const float bv = bias[n];
    __bf16* dst = (which == 0) ? qh : (which == 1) ? kh : vh;
#pragma unroll
    for (int r = 0; r < 8; ++r) {
      const int m = m0 + wv * 16 + lh * 8 + r;      // C tile row = 8*half + r
      const int s = m >> 2, b = m & 3;              // m = s*B + b
      const int bhd = b * H_ + h;
      float v = acc[c][r] + bv;
      if (which == 0) v *= qscale;
      dst[((size_t)bhd * S_ + s) * D_ + d] = (__bf16)v;
    }
  }
}

// ---------------------------------------------------------------------------
// Kernel 2: flash attention per head.  WG = 8 waves = 128 query rows.
//   Each wave owns a 16-row Q tile (2 resident A-fragments, D=64).
//   Per 32-key step: async-stage K/V block (32x64 bf16 each) into LDS,
//   4 WMMAs for QK^T, exp2-domain online softmax, P -> LDS -> A-fragment,
//   4 WMMAs for P*V.  Output written bf16 into [S*B][E] layout for kernel 3.
// ---------------------------------------------------------------------------
__global__ __launch_bounds__(256)
void mha_attn_kernel(const __bf16* __restrict__ qh, const __bf16* __restrict__ kh,
                     const __bf16* __restrict__ vh, __bf16* __restrict__ oh) {
  __shared__ __align__(16) __bf16 Kl[32][72];      // [key][d], 144B rows
  __shared__ __align__(16) __bf16 Vl[32][72];      // [key][d]
  __shared__ __align__(16) __bf16 Pl[8][16][40];   // per-wave P tile [row][key]

  const int tid = threadIdx.x;
  const int lane = tid & 31, wv = tid >> 5;
  const int l15 = lane & 15, lh = lane >> 4;
  const int bh = blockIdx.y;                       // b*H + h
  const int q0 = blockIdx.x * 128;

  const __bf16* Qb = qh + (size_t)bh * S_ * D_;
  const __bf16* Kb = kh + (size_t)bh * S_ * D_;
  const __bf16* Vb = vh + (size_t)bh * S_ * D_;

  // Resident Q fragments for this wave's 16 query rows.
  const int qrow = q0 + wv * 16 + l15;
  const __bf16* qp = Qb + (size_t)qrow * D_;
  v16bf qfrag[2];
#pragma unroll
  for (int kk = 0; kk < 2; ++kk)
    qfrag[kk] = make_frag(qp + kk * 32 + lh * 16, qp + kk * 32 + lh * 16 + 8);

  v8f o[4] = {};
  float mrow[8], lrow[8];
#pragma unroll
  for (int r = 0; r < 8; ++r) { mrow[r] = -1.0e30f; lrow[r] = 0.0f; }

  const int krow = tid >> 3, kpart = tid & 7;      // staging: 16B per thread

  for (int it = 0; it < S_ / 32; ++it) {
    const int key0 = it * 32;
    cp16_g2l(&Kl[krow][kpart * 8], Kb + (size_t)(key0 + krow) * D_ + kpart * 8);
    cp16_g2l(&Vl[krow][kpart * 8], Vb + (size_t)(key0 + krow) * D_ + kpart * 8);
    if (it + 1 < S_ / 32) {
      __builtin_prefetch(Kb + (size_t)(key0 + 32 + krow) * D_ + kpart * 8, 0, 1);
      __builtin_prefetch(Vb + (size_t)(key0 + 32 + krow) * D_ + kpart * 8, 0, 1);
    }
    cp_wait_all();
    __syncthreads();

    // scores: two 16x16 tiles, each = 2 WMMAs over D=64
    v8f sc[2];
#pragma unroll
    for (int t = 0; t < 2; ++t) {
      v16bf b0 = make_frag(&Kl[t * 16 + l15][lh * 16],
                           &Kl[t * 16 + l15][lh * 16 + 8]);
      v16bf b1 = make_frag(&Kl[t * 16 + l15][32 + lh * 16],
                           &Kl[t * 16 + l15][32 + lh * 16 + 8]);
      v8f z = {};
      v8f s1 = WMMA_BF16(qfrag[0], b0, z);
      sc[t]  = WMMA_BF16(qfrag[1], b1, s1);
    }

    // online softmax; rows striped: reg r holds row 8*lh+r, 16 lanes = 16 cols
    float p0[8], p1[8], corr[8];
#pragma unroll
    for (int r = 0; r < 8; ++r) {
      float mx = fmaxf(sc[0][r], sc[1][r]);
      mx = fmaxf(mx, __shfl_xor(mx, 1, 32));
      mx = fmaxf(mx, __shfl_xor(mx, 2, 32));
      mx = fmaxf(mx, __shfl_xor(mx, 4, 32));
      mx = fmaxf(mx, __shfl_xor(mx, 8, 32));
      const float nm = fmaxf(mrow[r], mx);
      p0[r] = exp2f(sc[0][r] - nm);
      p1[r] = exp2f(sc[1][r] - nm);
      float rs = p0[r] + p1[r];
      rs += __shfl_xor(rs, 1, 32);
      rs += __shfl_xor(rs, 2, 32);
      rs += __shfl_xor(rs, 4, 32);
      rs += __shfl_xor(rs, 8, 32);
      corr[r] = exp2f(mrow[r] - nm);
      lrow[r] = lrow[r] * corr[r] + rs;
      mrow[r] = nm;
    }
#pragma unroll
    for (int c = 0; c < 4; ++c)
#pragma unroll
      for (int r = 0; r < 8; ++r) o[c][r] *= corr[r];

    // P (C layout) -> LDS -> A-fragment re-layout (same-wave DS, in order)
#pragma unroll
    for (int r = 0; r < 8; ++r) {
      Pl[wv][lh * 8 + r][l15]      = (__bf16)p0[r];
      Pl[wv][lh * 8 + r][16 + l15] = (__bf16)p1[r];
    }
    v16bf pf = make_frag(&Pl[wv][l15][lh * 16], &Pl[wv][l15][lh * 16 + 8]);

    // O += P * V  (V B-fragments gathered column-wise from row-major Vl)
#pragma unroll
    for (int c = 0; c < 4; ++c) {
      v16bf bv;
#pragma unroll
      for (int i = 0; i < 16; ++i) bv[i] = Vl[lh * 16 + i][c * 16 + l15];
      o[c] = WMMA_BF16(pf, bv, o[c]);
    }
    __syncthreads();
  }

  // epilogue: normalize, write bf16 into [m = s*B + b][h*64 + d]
  const int b = bh >> 4, h = bh & 15;
#pragma unroll
  for (int r = 0; r < 8; ++r) {
    const float inv = 1.0f / lrow[r];
    const int s = q0 + wv * 16 + lh * 8 + r;
    const size_t base = (size_t)(s * B_ + b) * E_ + h * D_;
#pragma unroll
    for (int c = 0; c < 4; ++c)
      oh[base + c * 16 + l15] = (__bf16)(o[c][r] * inv);
  }
}

// ---------------------------------------------------------------------------
// Kernel 3: out = O @ Wout^T + bias.  M=8192, N=1024, K=1024, f32 output.
// ---------------------------------------------------------------------------
__global__ __launch_bounds__(256)
void mha_out_proj_kernel(const __bf16* __restrict__ A, const float* __restrict__ W,
                         const float* __restrict__ bias, float* __restrict__ out) {
  __shared__ __align__(16) __bf16 As[128][40];
  __shared__ __align__(16) __bf16 Bs[128][40];

  const int tid = threadIdx.x;
  const int lane = tid & 31, wv = tid >> 5;
  const int l15 = lane & 15, lh = lane >> 4;
  const int m0 = blockIdx.y * 128;
  const int n0 = blockIdx.x * 128;

  const int ldrow = tid >> 1;
  const int ldh   = tid & 1;
  const __bf16* asrc = A + (size_t)(m0 + ldrow) * E_ + ldh * 16;
  const float*  wsrc = W + (size_t)(n0 + ldrow) * E_ + ldh * 16;

  v8f acc[8] = {};

  for (int kb = 0; kb < E_; kb += 32) {
    *(v8bf*)&As[ldrow][ldh * 16]     = ((const v8bf*)asrc)[0];
    *(v8bf*)&As[ldrow][ldh * 16 + 8] = ((const v8bf*)asrc)[1];
    cvt16_f32_to_bf16(&Bs[ldrow][ldh * 16], (const float4*)wsrc);
    asrc += 32; wsrc += 32;
    if (kb + 32 < E_) {
      __builtin_prefetch(asrc, 0, 1);
      __builtin_prefetch(wsrc, 0, 1);
    }
    __syncthreads();

    v16bf afrag = make_frag(&As[wv * 16 + l15][lh * 16],
                            &As[wv * 16 + l15][lh * 16 + 8]);
#pragma unroll
    for (int c = 0; c < 8; ++c) {
      v16bf bfrag = make_frag(&Bs[c * 16 + l15][lh * 16],
                              &Bs[c * 16 + l15][lh * 16 + 8]);
      acc[c] = WMMA_BF16(afrag, bfrag, acc[c]);
    }
    __syncthreads();
  }

#pragma unroll
  for (int c = 0; c < 8; ++c) {
    const int n = n0 + c * 16 + l15;
    const float bv = bias[n];
#pragma unroll
    for (int r = 0; r < 8; ++r) {
      const int m = m0 + wv * 16 + lh * 8 + r;
      out[(size_t)m * E_ + n] = acc[c][r] + bv;
    }
  }
}

// ---------------------------------------------------------------------------
extern "C" void kernel_launch(void* const* d_in, const int* in_sizes, int n_in,
                              void* d_out, int out_size, void* d_ws, size_t ws_size,
                              hipStream_t stream) {
  (void)in_sizes; (void)n_in; (void)out_size; (void)ws_size;  // needs 64 MB ws
  const float* x     = (const float*)d_in[0];
  const float* w_in  = (const float*)d_in[1];
  const float* b_in  = (const float*)d_in[2];
  const float* w_out = (const float*)d_in[3];
  const float* b_out = (const float*)d_in[4];
  float* out = (float*)d_out;

  __bf16* qh = (__bf16*)d_ws;
  __bf16* kh = qh + HEAD_ELEMS;
  __bf16* vh = kh + HEAD_ELEMS;
  __bf16* oh = vh + HEAD_ELEMS;

  // 1) QKV projection: tiles (3072/128) x (8192/128)
  mha_qkv_proj_kernel<<<dim3(N_QKV / 128, (S_ * B_) / 128), 256, 0, stream>>>(
      x, w_in, b_in, qh, kh, vh);
  // 2) Attention: 16 query blocks x 64 heads
  mha_attn_kernel<<<dim3(S_ / 128, B_ * H_), 256, 0, stream>>>(qh, kh, vh, oh);
  // 3) Output projection: (1024/128) x (8192/128)
  mha_out_proj_kernel<<<dim3(E_ / 128, (S_ * B_) / 128), 256, 0, stream>>>(
      oh, w_out, b_out, out);
}
